// Attention_13597866459364
// MI455X (gfx1250) — compile-verified, matrix operand order checked by reading, not run on hardware
//
#include <hip/hip_runtime.h>
#include <hip/hip_bf16.h>

// ---- problem constants (match reference) ----
constexpr int Bb = 8, Nn = 1024, Cc = 768, Hh = 12, HD = 64;
constexpr int BH = Bb * Hh;              // 96
constexpr int Mrows = Bb * Nn;           // 8192
constexpr float SCALE = 0.125f;          // 64^-0.5
constexpr unsigned RANK = (unsigned)((size_t)Nn * Nn / 2);   // 524288 (0-based desc idx)
constexpr size_t ATTN_ELEMS = (size_t)BH * Nn * Nn;          // 100663296
constexpr size_t ATTN_OFF = (size_t)Mrows * Cc;              // out segment size

typedef __attribute__((ext_vector_type(16))) __bf16 v16bf;
typedef __attribute__((ext_vector_type(8)))  float  v8f;
typedef int v4i __attribute__((__vector_size__(16)));        // matches builtin proto

#define DI __device__ __forceinline__

// CDNA5 async global->LDS copy path (ASYNCcnt), guarded so the file compiles
// on toolchains lacking the builtins.
#if defined(__HIP_DEVICE_COMPILE__) &&                                         \
    __has_builtin(__builtin_amdgcn_global_load_async_to_lds_b128) &&           \
    __has_builtin(__builtin_amdgcn_s_wait_asynccnt)
#define USE_ASYNC_LDS 1
#else
#define USE_ASYNC_LDS 0
#endif

union BFrag { v16bf v; unsigned short u[16]; };

DI unsigned short f2bf(float f) {                    // round-to-nearest-even f32->bf16
  unsigned u = __float_as_uint(f);
  u += 0x7FFFu + ((u >> 16) & 1u);
  return (unsigned short)(u >> 16);
}

// A-style fragment (16x32 bf16): lane holds row (lane&15); half=lane>>4 selects the
// K sub-block. Per lane this is two 32B runs -> four float4 loads (b128).
// tile -> element [row0][k0]; ld = row stride in floats. Works for global or LDS.
DI v16bf load_frag_row(const float* __restrict__ tile, int ld) {
  int lane = threadIdx.x & 31;
  const float* r0 = tile + (size_t)(lane & 15) * ld + ((lane >> 4) << 3);
  float4 q0 = *(const float4*)(r0);        // kb 0..3
  float4 q1 = *(const float4*)(r0 + 4);    // kb 4..7
  float4 q2 = *(const float4*)(r0 + 16);   // kb 16..19
  float4 q3 = *(const float4*)(r0 + 20);   // kb 20..23
  BFrag f;
  f.u[0]  = f2bf(q0.x); f.u[1]  = f2bf(q0.y); f.u[2]  = f2bf(q0.z); f.u[3]  = f2bf(q0.w);
  f.u[4]  = f2bf(q1.x); f.u[5]  = f2bf(q1.y); f.u[6]  = f2bf(q1.z); f.u[7]  = f2bf(q1.w);
  f.u[8]  = f2bf(q2.x); f.u[9]  = f2bf(q2.y); f.u[10] = f2bf(q2.z); f.u[11] = f2bf(q2.w);
  f.u[12] = f2bf(q3.x); f.u[13] = f2bf(q3.y); f.u[14] = f2bf(q3.z); f.u[15] = f2bf(q3.w);
  return f.v;
}

// B fragment from a K-major matrix (row index = k): element [k][n], n = lane&15.
DI v16bf load_frag_col(const float* __restrict__ tile, int ld) {
  int lane = threadIdx.x & 31;
  int n = lane & 15;
  int kh = (lane >> 4) << 3;
  BFrag f;
#pragma unroll
  for (int r = 0; r < 8; ++r) {
    int kb = ((r < 4) ? (r << 1) : (16 + ((r - 4) << 1))) + kh;
    f.u[2 * r]     = f2bf(tile[(size_t)kb * ld + n]);
    f.u[2 * r + 1] = f2bf(tile[(size_t)(kb + 1) * ld + n]);
  }
  return f.v;
}

DI v8f wmma_bf16(v16bf a, v16bf b, v8f c) {
  return __builtin_amdgcn_wmma_f32_16x16x32_bf16(false, a, false, b, (short)0, c,
                                                 false, false);
}

// ---------------- kernel 1: qkv projection ----------------
__global__ __launch_bounds__(256) void qkv_kernel(const float* __restrict__ x,
                                                  const float* __restrict__ wqkv,
                                                  float* __restrict__ q,
                                                  float* __restrict__ k,
                                                  float* __restrict__ v) {
  constexpr int NT = 3 * Cc / 16;                       // 144 col tiles
  int wid = blockIdx.x * (blockDim.x >> 5) + (threadIdx.x >> 5);
  int tm = wid / NT, tn = wid % NT;
  const float* arow = x + (size_t)tm * 16 * Cc;
  const float* brow = wqkv + (size_t)tn * 16 * Cc;
  v8f acc = {};
  for (int k0 = 0; k0 < Cc; k0 += 32) {
    if (k0 + 32 < Cc) {
      __builtin_prefetch(arow + k0 + 32, 0, 3);
      __builtin_prefetch(brow + k0 + 32, 0, 3);
    }
    v16bf a = load_frag_row(arow + k0, Cc);
    v16bf b = load_frag_row(brow + k0, Cc);
    acc = wmma_bf16(a, b, acc);
  }
  int lane = threadIdx.x & 31;
  int col = tn * 16 + (lane & 15);                      // 0..2303
  int rbase = (lane >> 4) << 3;
  int which = col / Cc;                                 // 0=q 1=k 2=v
  int cj = col % Cc;
  int h = cj / HD, d = cj % HD;
  float* dst = (which == 0) ? q : ((which == 1) ? k : v);
  float s = (which == 0) ? SCALE : 1.0f;
#pragma unroll
  for (int r = 0; r < 8; ++r) {
    int m = tm * 16 + rbase + r;
    int b_ = m / Nn, n_ = m % Nn;
    dst[(((size_t)(b_ * Hh + h) * Nn + n_) * HD + d)] = acc[r] * s;
  }
}

// ------- fused kernel: scores (q@k^T + log bias) + row softmax + coarse hist -------
// One block owns a 16-row x 1024-col panel of attn for one (b,h). 8 waves x 8 tiles
// x 2 WMMA. Panel lives in LDS (64KB of the 320KB WGP LDS), written to HBM once.
__global__ __launch_bounds__(256) void scores_softmax_kernel(
    const float* __restrict__ q, const float* __restrict__ kmat,
    const float* __restrict__ tok, float* __restrict__ attn,
    unsigned* __restrict__ hist) {
  __shared__ float panel[16 * Nn];          // 64 KB
  __shared__ float logtok[Nn];              // 4 KB
  __shared__ unsigned lh[1024];             // 4 KB
  int bh = blockIdx.x >> 6;                 // / (Nn/16)
  int tq = blockIdx.x & 63;
  int tid = threadIdx.x;
  int b_ = bh / Hh;
  for (int i = tid; i < 1024; i += 256) {
    lh[i] = 0u;
    logtok[i] = __logf(tok[b_ * Nn + i]);
  }
  __syncthreads();

  int w = tid >> 5;
  int lane = tid & 31;
  int n = lane & 15;
  int rbase = (lane >> 4) << 3;
  const float* qp = q + (size_t)bh * Nn * HD + (size_t)tq * 16 * HD;
  v16bf a0 = load_frag_row(qp, HD);
  v16bf a1 = load_frag_row(qp + 32, HD);
#pragma unroll 1
  for (int i = 0; i < 8; ++i) {
    int tk = (i << 3) + w;
    const float* kp = kmat + (size_t)bh * Nn * HD + (size_t)tk * 16 * HD;
    v8f acc = {};
    acc = wmma_bf16(a0, load_frag_row(kp, HD), acc);
    acc = wmma_bf16(a1, load_frag_row(kp + 32, HD), acc);
    int col = tk * 16 + n;
    float lb = logtok[col];
#pragma unroll
    for (int r = 0; r < 8; ++r)
      panel[(rbase + r) * Nn + col] = acc[r] + lb;
  }
  __syncthreads();

  // softmax: 16 threads per row (width-16 wave32 shuffles)
  int g = tid >> 4;                          // row 0..15
  int t16 = tid & 15;
  float* row = panel + g * Nn;
  float m = -3.0e38f;
  for (int c = t16; c < Nn; c += 16) m = fmaxf(m, row[c]);
#pragma unroll
  for (int off = 8; off; off >>= 1) m = fmaxf(m, __shfl_xor(m, off, 16));
  float s = 0.f;
  for (int c = t16; c < Nn; c += 16) {
    float e = __expf(row[c] - m);
    row[c] = e;
    s += e;
  }
#pragma unroll
  for (int off = 8; off; off >>= 1) s += __shfl_xor(s, off, 16);
  float inv = 1.0f / s;
  for (int c = t16; c < Nn; c += 16) {
    float p = row[c] * inv;
    row[c] = p;
    unsigned bkt = __float_as_uint(p) >> 20;   // monotone for p in [0,1]
    if (bkt > 1023u) bkt = 1023u;
    atomicAdd(&lh[bkt], 1u);
  }
  __syncthreads();

  float4* dst = (float4*)(attn + (size_t)bh * Nn * Nn + (size_t)tq * 16 * Nn);
  const float4* src = (const float4*)panel;
  for (int i = tid; i < 16 * Nn / 4; i += 256) dst[i] = src[i];
  for (int i = tid; i < 1024; i += 256)
    if (lh[i]) atomicAdd(&hist[bh * 1024 + i], lh[i]);
}

// ---------------- top-k threshold: two-pass radix select on float bits ----------------
__global__ void zero_kernel(unsigned* h, int n) {
  int i = blockIdx.x * 256 + threadIdx.x;
  if (i < n) h[i] = 0;
}

constexpr int HBLK = 64;                                // blocks per (b,h) slice

__global__ void select1_kernel(const unsigned* __restrict__ hist,
                               int* __restrict__ selb, unsigned* __restrict__ selr) {
  int bh = threadIdx.x;
  if (bh >= BH) return;
  const unsigned* h = hist + bh * 1024;
  unsigned cum = 0;
  int b = 1023;
  for (; b > 0; --b) {
    unsigned c = h[b];
    if (cum + c > RANK) break;
    cum += c;
  }
  selb[bh] = b;
  selr[bh] = RANK - cum;
}

__global__ __launch_bounds__(256) void hist2_kernel(const float* __restrict__ attn,
                                                    const int* __restrict__ selb,
                                                    unsigned* __restrict__ hist2) {
  __shared__ unsigned lh[1024];
  int bh = blockIdx.y;
  int sb = selb[bh];
  for (int i = threadIdx.x; i < 1024; i += 256) lh[i] = 0;
  __syncthreads();
  const float4* base =
      (const float4*)(attn + (size_t)bh * Nn * Nn + (size_t)blockIdx.x * ((Nn * Nn) / HBLK));
  constexpr int per4 = (Nn * Nn) / HBLK / 4;            // 4096 float4 per block
  for (int i = threadIdx.x; i < per4; i += 256) {
    float4 vb = base[i];
    float pv[4] = {vb.x, vb.y, vb.z, vb.w};
#pragma unroll
    for (int j = 0; j < 4; ++j) {
      unsigned bits = __float_as_uint(pv[j]);
      unsigned bkt = bits >> 20;
      if (bkt > 1023u) bkt = 1023u;
      if ((int)bkt == sb) atomicAdd(&lh[(bits >> 10) & 1023u], 1u);
    }
  }
  __syncthreads();
  for (int i = threadIdx.x; i < 1024; i += 256)
    if (lh[i]) atomicAdd(&hist2[bh * 1024 + i], lh[i]);
}

__global__ void select2_kernel(const unsigned* __restrict__ hist2,
                               const int* __restrict__ selb,
                               const unsigned* __restrict__ selr,
                               float* __restrict__ sigma) {
  int bh = threadIdx.x;
  if (bh >= BH) return;
  const unsigned* h = hist2 + bh * 1024;
  unsigned rank = selr[bh], cum = 0;
  int s = 1023;
  for (; s > 0; --s) {
    unsigned c = h[s];
    if (cum + c > rank) break;
    cum += c;
  }
  unsigned bitsLow = ((unsigned)selb[bh] << 20) | ((unsigned)s << 10);
  sigma[bh] = __uint_as_float(bitsLow);
}

__global__ __launch_bounds__(256) void sparsify_kernel(float* __restrict__ attn,
                                                       const float* __restrict__ sigma) {
  size_t i4 = (size_t)blockIdx.x * 256 + threadIdx.x;
  int bh = (int)((i4 * 4) / ((size_t)Nn * Nn));
  float sg = sigma[bh];
  float4* p = reinterpret_cast<float4*>(attn) + i4;
  float4 vb = *p;
  vb.x = (vb.x >= sg) ? vb.x : 0.f;
  vb.y = (vb.y >= sg) ? vb.y : 0.f;
  vb.z = (vb.z >= sg) ? vb.z : 0.f;
  vb.w = (vb.w >= sg) ? vb.w : 0.f;
  *p = vb;
}

// ---------------- kernel: ctx = attn @ v, written as [B,N,H*HD] ----------------
// V tiles (32x16) are staged into LDS with async global->LDS copies, ping-pong
// buffered, pipelined with s_wait_asynccnt (CDNA5 ASYNCcnt path).
__global__ __launch_bounds__(256) void ctx_kernel(const float* __restrict__ attn,
                                                  const float* __restrict__ v,
                                                  float* __restrict__ ctx) {
  constexpr int TPB = (Nn / 16) * (HD / 16);            // 256 tiles per (b,h)
  int wid = blockIdx.x * (blockDim.x >> 5) + (threadIdx.x >> 5);
  int bh = wid / TPB;
  int t = wid % TPB;
  int tq = t / (HD / 16), td = t % (HD / 16);
  const float* ap = attn + (size_t)bh * Nn * Nn + (size_t)tq * 16 * Nn;
  const float* vp = v + (size_t)bh * Nn * HD + td * 16;
  int w = threadIdx.x >> 5, lane = threadIdx.x & 31;
  v8f acc = {};
#if USE_ASYNC_LDS
  __shared__ float vstage[8][2][32 * 16];               // 32 KB, per-wave ping-pong
  auto stage = [&](int buf, int k0) {
#pragma unroll
    for (int i = 0; i < 4; ++i) {                       // 4 x b128 covers 32x16 f32
      int idx = i * 32 + lane;                          // float4 index
      const float* g = vp + (size_t)(k0 + (idx >> 2)) * HD + ((idx & 3) << 2);
      float* l = &vstage[w][buf][idx << 2];
      __builtin_amdgcn_global_load_async_to_lds_b128(
          (__attribute__((address_space(1))) v4i*)g,
          (__attribute__((address_space(3))) v4i*)l, 0, 0);
    }
  };
  stage(0, 0);
  for (int k0 = 0; k0 < Nn; k0 += 32) {
    int cur = (k0 >> 5) & 1;
    if (k0 + 32 < Nn) {
      stage(cur ^ 1, k0 + 32);                          // prefetch next tile
      __builtin_amdgcn_s_wait_asynccnt(4);              // FIFO: first 4 (cur) done
    } else {
      __builtin_amdgcn_s_wait_asynccnt(0);
    }
    v16bf a = load_frag_row(ap + k0, Nn);
    v16bf b = load_frag_col(&vstage[w][cur][0], 16);
    acc = wmma_bf16(a, b, acc);
  }
#else
  for (int k0 = 0; k0 < Nn; k0 += 32) {
    v16bf a = load_frag_row(ap + k0, Nn);
    v16bf b = load_frag_col(vp + (size_t)k0 * HD, HD);
    acc = wmma_bf16(a, b, acc);
  }
#endif
  int b_ = bh / Hh, h_ = bh % Hh;
  int d = td * 16 + (lane & 15);
  int q0 = tq * 16 + ((lane >> 4) << 3);
#pragma unroll
  for (int r = 0; r < 8; ++r)
    ctx[(size_t)(b_ * Nn + q0 + r) * Cc + h_ * HD + d] = acc[r];
}

// ---------------- kernel: out = ctx @ w_proj^T + b_proj ----------------
__global__ __launch_bounds__(256) void proj_kernel(const float* __restrict__ ctx,
                                                   const float* __restrict__ wp,
                                                   const float* __restrict__ bp,
                                                   float* __restrict__ out) {
  constexpr int NT = Cc / 16;                           // 48
  int wid = blockIdx.x * (blockDim.x >> 5) + (threadIdx.x >> 5);
  int tm = wid / NT, tn = wid % NT;
  const float* arow = ctx + (size_t)tm * 16 * Cc;
  const float* brow = wp + (size_t)tn * 16 * Cc;
  v8f acc = {};
  for (int k0 = 0; k0 < Cc; k0 += 32) {
    if (k0 + 32 < Cc) {
      __builtin_prefetch(arow + k0 + 32, 0, 3);
      __builtin_prefetch(brow + k0 + 32, 0, 3);
    }
    v16bf a = load_frag_row(arow + k0, Cc);
    v16bf b = load_frag_row(brow + k0, Cc);
    acc = wmma_bf16(a, b, acc);
  }
  int lane = threadIdx.x & 31;
  int col = tn * 16 + (lane & 15);
  float bias = bp[col];
  int r0 = tm * 16 + ((lane >> 4) << 3);
#pragma unroll
  for (int r = 0; r < 8; ++r) out[(size_t)(r0 + r) * Cc + col] = acc[r] + bias;
}

extern "C" void kernel_launch(void* const* d_in, const int* in_sizes, int n_in,
                              void* d_out, int out_size, void* d_ws, size_t ws_size,
                              hipStream_t stream) {
  (void)in_sizes; (void)n_in; (void)out_size; (void)ws_size;
  const float* x    = (const float*)d_in[0];
  const float* tok  = (const float*)d_in[1];
  const float* wqkv = (const float*)d_in[2];
  const float* wp   = (const float*)d_in[3];
  const float* bp   = (const float*)d_in[4];
  float* out  = (float*)d_out;
  float* attn = out + ATTN_OFF;

  char* w = (char*)d_ws;
  const size_t qkvN = (size_t)BH * Nn * HD;
  float* qb  = (float*)w; w += qkvN * 4;
  float* kb  = (float*)w; w += qkvN * 4;
  float* vb  = (float*)w; w += qkvN * 4;
  float* ctx = (float*)w; w += (size_t)Mrows * Cc * 4;
  unsigned* h1 = (unsigned*)w; w += (size_t)BH * 1024 * 4;
  unsigned* h2 = (unsigned*)w; w += (size_t)BH * 1024 * 4;
  int*      selb = (int*)w;      w += BH * 4;
  unsigned* selr = (unsigned*)w; w += BH * 4;
  float*    sigma = (float*)w;   w += BH * 4;

  // h1 and h2 are contiguous: zero both (must be re-zeroed every call).
  zero_kernel<<<(2 * BH * 1024 + 255) / 256, 256, 0, stream>>>(h1, 2 * BH * 1024);

  qkv_kernel<<<(Mrows / 16) * (3 * Cc / 16) / 8, 256, 0, stream>>>(x, wqkv, qb, kb, vb);
  scores_softmax_kernel<<<BH * (Nn / 16), 256, 0, stream>>>(qb, kb, tok, attn, h1);

  select1_kernel<<<1, 128, 0, stream>>>(h1, selb, selr);
  dim3 hgrid(HBLK, BH);
  hist2_kernel<<<hgrid, 256, 0, stream>>>(attn, selb, h2);
  select2_kernel<<<1, 128, 0, stream>>>(h2, selb, selr, sigma);
  sparsify_kernel<<<(unsigned)(ATTN_ELEMS / 4 / 256), 256, 0, stream>>>(attn, sigma);

  ctx_kernel<<<BH * (Nn / 16) * (HD / 16) / 8, 256, 0, stream>>>(attn, vb, ctx);
  proj_kernel<<<(Mrows / 16) * (Cc / 16) / 8, 256, 0, stream>>>(ctx, wp, bp, out);
}
